// TGAT_Encoder_69002944578160
// MI455X (gfx1250) — compile-verified, matrix operand order checked by reading, not run on hardware
//
#include <hip/hip_runtime.h>
#include <hip/hip_bf16.h>
#include <cstdint>
#include <cstddef>

// ---------------------------------------------------------------------------
// TGAT encoder for MI455X (gfx1250, wave32, WMMA).
// All dense GEMMs (Q/K/V proj, fc, agg MLP) run through
// v_wmma_f32_16x16x32_bf16. A strips are staged in LDS (shared by the 8
// waves of a block), B (weights) stream from global/L2, 2 M-tiles + up to
// 2 N-outputs per wave for register-level reuse.
// ---------------------------------------------------------------------------

#define FEAT      172
#define MODEL     516          // 3*FEAT
#define NHEAD     2
#define DKH       258          // MODEL / NHEAD
#define NNEI      16
#define BATCH     512
#define KP_QKV    544          // 516 padded to 17*32
#define NT_QKV    33           // 528/16 output tiles for 516-wide outputs
#define NP_QKV    528
#define KP_AGG1   704          // 688 padded to 22*32
#define NT_AGG    11           // 176/16
#define NP_AGG    176
#define KP_AGG2   192          // 172 padded to 6*32
#define CHUNK     512          // rows per pipeline chunk (divides 512 and 8192)
#define KP_MAX    704
#define LDS_PAD   8            // halves of per-row pad to stagger LDS banks

typedef __bf16 bf16_t;
typedef __attribute__((ext_vector_type(16))) __bf16 v16bf;
typedef __attribute__((ext_vector_type(8)))  __bf16 v8bf;
typedef __attribute__((ext_vector_type(8)))  float  v8f;

static __device__ __forceinline__ v16bf pack16(v8bf lo, v8bf hi) {
  return __builtin_shufflevector(lo, hi, 0,1,2,3,4,5,6,7,8,9,10,11,12,13,14,15);
}
static __device__ __forceinline__ v8f wmma_bf16(v16bf a, v16bf b, v8f c) {
  return __builtin_amdgcn_wmma_f32_16x16x32_bf16(false, a, false, b, (short)0, c,
                                                 false, false);
}

// --------------------------- weight packing --------------------------------
__global__ void k_pack_weight(const float* __restrict__ W, bf16_t* __restrict__ Wb,
                              int rows, int cols, int rowsP, int colsP) {
  int c = blockIdx.x * blockDim.x + threadIdx.x;
  int r = blockIdx.y;
  if (c >= colsP || r >= rowsP) return;
  float v = (r < rows && c < cols) ? W[(size_t)r * cols + c] : 0.0f;
  Wb[(size_t)r * colsP + c] = (bf16_t)v;
}

__global__ void k_zero_bf16(bf16_t* __restrict__ p, size_t n) {
  size_t i = (size_t)blockIdx.x * blockDim.x + threadIdx.x;
  if (i < n) p[i] = (bf16_t)0.0f;
}

// flatten one level of neighbor expansion: row_nodes/row_ts for 512*16 rows
__global__ void k_gather_rows(const int* __restrict__ idx,
                              const int* __restrict__ ngh_node_tab,
                              const float* __restrict__ ngh_ts_tab,
                              int* __restrict__ out_nodes, float* __restrict__ out_ts,
                              int B) {
  int t = blockIdx.x * blockDim.x + threadIdx.x;
  if (t >= B * NNEI) return;
  int b = t / NNEI, j = t % NNEI;
  int node = idx[b];
  out_nodes[t] = ngh_node_tab[(size_t)node * NNEI + j];
  out_ts[t]    = ngh_ts_tab[(size_t)node * NNEI + j];
}

// ----------------------- activation matrix builders ------------------------
// K/V input row (per row r, neighbor j): [ngh_conv(172) | e_feat(172) | cos-time(172) | 0 pad]
__global__ void k_build_kv(const int* __restrict__ row_nodes, const float* __restrict__ row_ts,
                           int c0,
                           const int* __restrict__ ngh_node_tab,
                           const int* __restrict__ ngh_eidx_tab,
                           const float* __restrict__ ngh_ts_tab,
                           const float* __restrict__ n_feat, const float* __restrict__ e_feat,
                           const float* __restrict__ basis_freq, const float* __restrict__ phase,
                           const float* __restrict__ conv_rows, int conv_gather,
                           bf16_t* __restrict__ Kin) {
  int rj = blockIdx.x;          // chunk-local row*16+neighbor
  int d  = threadIdx.x;         // 0..543
  int r = rj >> 4, j = rj & 15;
  int node = row_nodes[c0 + r];
  float val = 0.0f;
  if (d < FEAT) {
    if (conv_gather) {
      int nid = ngh_node_tab[(size_t)node * NNEI + j];
      val = n_feat[(size_t)nid * FEAT + d];
    } else {
      val = conv_rows[((size_t)(c0 + r) * NNEI + j) * FEAT + d];
    }
  } else if (d < 2 * FEAT) {
    int e = ngh_eidx_tab[(size_t)node * NNEI + j];
    val = e_feat[(size_t)e * FEAT + (d - FEAT)];
  } else if (d < 3 * FEAT) {
    int f = d - 2 * FEAT;
    float dt = row_ts[c0 + r] - ngh_ts_tab[(size_t)node * NNEI + j];
    val = cosf(dt * basis_freq[f] + phase[f]);
  }
  Kin[(size_t)rj * KP_QKV + d] = (bf16_t)val;
}

// Q input row: [src_conv(172) | zeros(172) | cos(phase)(172) | 0 pad]
__global__ void k_build_q(const float* __restrict__ src_conv, int src_gather,
                          const int* __restrict__ row_nodes,
                          const float* __restrict__ n_feat, const float* __restrict__ phase,
                          int c0, bf16_t* __restrict__ Qin) {
  int r = blockIdx.x;
  int d = threadIdx.x;
  float val = 0.0f;
  if (d < FEAT) {
    val = src_gather ? n_feat[(size_t)row_nodes[c0 + r] * FEAT + d]
                     : src_conv[(size_t)(c0 + r) * FEAT + d];
  } else if (d >= 2 * FEAT && d < 3 * FEAT) {
    val = cosf(phase[d - 2 * FEAT]);   // time_encode(0)
  }
  Qin[(size_t)r * KP_QKV + d] = (bf16_t)val;
}

// ------------------------------ WMMA GEMM ----------------------------------
// D[m][n] = sum_k A[m][k] * Bw[n][k]   (A: M x Kp bf16, Bw: Npad x Kp bf16)
// Block = 256 threads = 8 waves. All waves share one 32-row A strip staged
// in LDS; each wave owns one 16-wide N tile and 2 M tiles (4 WMMAs/K-step in
// the dual-weight K/V variant). M must be a multiple of 32 (all stages are).
// Fragment layout per CDNA5 ISA 7.12.2 (16-bit A 16x32): lane<16 holds
// K {kc..kc+7, kc+16..kc+23} of row m0+lane, lane>=16 the other K halves.
template<bool DUAL>
__global__ __launch_bounds__(256)
void k_gemm_bf16(const bf16_t* __restrict__ A, int lda, int M,
                 const bf16_t* __restrict__ B1, const bf16_t* __restrict__ B2,
                 int Kp, int nTiles,
                 float* __restrict__ D1, float* __restrict__ D2, int ldd, int Nstore,
                 bf16_t* __restrict__ Dbf, int lddbf, int relu) {
  __shared__ bf16_t As[32 * (KP_MAX + LDS_PAD)];
  const int lds = Kp + LDS_PAD;
  const int m0 = blockIdx.x * 32;
  const int t = threadIdx.x;

  // cooperative global -> LDS stage of the 32-row A strip (b128 chunks)
  const int chunksPerRow = Kp >> 3;
  for (int c = t; c < 32 * chunksPerRow; c += 256) {
    int row = c / chunksPerRow;
    int col = (c - row * chunksPerRow) << 3;
    *(v8bf*)(As + row * lds + col) = *(const v8bf*)(A + (size_t)(m0 + row) * lda + col);
  }
  __syncthreads();

  const int wave = t >> 5;
  const int lane = t & 31;
  const int tn = blockIdx.y * 8 + wave;
  if (tn < nTiles) {                 // wave-uniform: EXEC stays all-ones
    const int n0 = tn * 16;
    const int sel = lane >> 4;       // which K-half this lane owns
    const int l   = lane & 15;
    const bf16_t* a0 = As + l * lds + sel * 8;          // M tile 0
    const bf16_t* a1 = As + (16 + l) * lds + sel * 8;   // M tile 1
    const bf16_t* br1 = B1 + (size_t)(n0 + l) * Kp + sel * 8;
    const bf16_t* br2 = DUAL ? (B2 + (size_t)(n0 + l) * Kp + sel * 8) : nullptr;
    v8f acc00 = {}, acc10 = {}, acc01 = {}, acc11 = {};
    for (int kc = 0; kc < Kp; kc += 32) {
      v16bf af0 = pack16(*(const v8bf*)(a0 + kc), *(const v8bf*)(a0 + kc + 16));
      v16bf af1 = pack16(*(const v8bf*)(a1 + kc), *(const v8bf*)(a1 + kc + 16));
      v16bf b1  = pack16(*(const v8bf*)(br1 + kc), *(const v8bf*)(br1 + kc + 16));
      acc00 = wmma_bf16(af0, b1, acc00);
      acc10 = wmma_bf16(af1, b1, acc10);
      if (DUAL) {
        v16bf b2 = pack16(*(const v8bf*)(br2 + kc), *(const v8bf*)(br2 + kc + 16));
        acc01 = wmma_bf16(af0, b2, acc01);
        acc11 = wmma_bf16(af1, b2, acc11);
      }
    }
    // C/D layout: lane<16 -> N=lane, M rows m0+{0..7}; lane>=16 -> m0+{8..15}
    const int n = n0 + l;
    if (n < Nstore) {
      for (int mt = 0; mt < 2; ++mt) {
        v8f accP = mt ? acc10 : acc00;
        v8f accS = mt ? acc11 : acc01;
        for (int v = 0; v < 8; ++v) {
          int m = m0 + mt * 16 + sel * 8 + v;
          if (m < M) {
            float x = accP[v];
            if (relu) x = fmaxf(x, 0.0f);
            if (D1)  D1[(size_t)m * ldd + n]    = x;
            if (Dbf) Dbf[(size_t)m * lddbf + n] = (bf16_t)x;
            if (DUAL && D2) D2[(size_t)m * ldd + n] = accS[v];
          }
        }
      }
    }
  }
}

// --------------------------- attention epilogue ----------------------------
__global__ __launch_bounds__(128)
void k_attention(const float* __restrict__ qh, const float* __restrict__ kh,
                 const float* __restrict__ vh,
                 const int* __restrict__ row_nodes, int c0,
                 const int* __restrict__ ngh_node_tab,
                 bf16_t* __restrict__ Obf) {
  __shared__ float s_w[NHEAD][NNEI];
  int r = blockIdx.x;
  int t = threadIdx.x;
  if (t < NHEAD * NNEI) {            // 32 (head, neighbor) score dots
    int h = t >> 4, j = t & 15;
    const float* qq = qh + (size_t)r * MODEL + h * DKH;
    const float* kk = kh + ((size_t)r * NNEI + j) * MODEL + h * DKH;
    float s = 0.0f;
    for (int d = 0; d < DKH; ++d) s += qq[d] * kk[d];
    s /= sqrtf((float)DKH);
    int node = row_nodes[c0 + r];
    if (ngh_node_tab[(size_t)node * NNEI + j] == 0) s = -1e10f;
    s_w[h][j] = s;
  }
  __syncthreads();
  if (t < NHEAD) {                   // softmax per head
    float mx = -3.0e38f;
    for (int j = 0; j < NNEI; ++j) mx = fmaxf(mx, s_w[t][j]);
    float sum = 0.0f;
    for (int j = 0; j < NNEI; ++j) { float e = expf(s_w[t][j] - mx); s_w[t][j] = e; sum += e; }
    float inv = 1.0f / sum;
    for (int j = 0; j < NNEI; ++j) s_w[t][j] *= inv;
  }
  __syncthreads();
  for (int d = t; d < KP_QKV; d += 128) {   // weighted V + bf16 pad for fc GEMM
    float o = 0.0f;
    if (d < MODEL) {
      int h = d / DKH;
      const float* vcol = vh + (size_t)r * NNEI * MODEL + d;
      for (int j = 0; j < NNEI; ++j) o += s_w[h][j] * vcol[(size_t)j * MODEL];
    }
    Obf[(size_t)r * KP_QKV + d] = (bf16_t)o;
  }
}

// residual + LayerNorm + assemble 688-wide MLP input (padded bf16 704)
__global__ __launch_bounds__(256)
void k_ln_agg(const float* __restrict__ ofc,
              const float* __restrict__ src_conv, int src_gather,
              const int* __restrict__ row_nodes, const float* __restrict__ n_feat,
              const float* __restrict__ phase,
              const float* __restrict__ gamma, const float* __restrict__ beta,
              int c0, bf16_t* __restrict__ AggIn) {
  __shared__ float red[256];
  __shared__ float s_y[MODEL];
  int r = blockIdx.x, t = threadIdx.x;
  float part = 0.0f;
  for (int d = t; d < MODEL; d += 256) {
    float qv;
    if (d < FEAT)
      qv = src_gather ? n_feat[(size_t)row_nodes[c0 + r] * FEAT + d]
                      : src_conv[(size_t)(c0 + r) * FEAT + d];
    else if (d < 2 * FEAT) qv = 0.0f;
    else qv = cosf(phase[d - 2 * FEAT]);
    float y = ofc[(size_t)r * MODEL + d] + qv;
    s_y[d] = y; part += y;
  }
  red[t] = part; __syncthreads();
  for (int s2 = 128; s2 > 0; s2 >>= 1) { if (t < s2) red[t] += red[t + s2]; __syncthreads(); }
  float mu = red[0] / (float)MODEL;
  __syncthreads();
  float vpart = 0.0f;
  for (int d = t; d < MODEL; d += 256) { float z = s_y[d] - mu; vpart += z * z; }
  red[t] = vpart; __syncthreads();
  for (int s2 = 128; s2 > 0; s2 >>= 1) { if (t < s2) red[t] += red[t + s2]; __syncthreads(); }
  float rstd = rsqrtf(red[0] / (float)MODEL + 1e-5f);
  for (int d = t; d < KP_AGG1; d += 256) {
    float val = 0.0f;
    if (d < MODEL) {
      val = (s_y[d] - mu) * rstd * gamma[d] + beta[d];
    } else if (d < MODEL + FEAT) {
      int dd = d - MODEL;
      val = src_gather ? n_feat[(size_t)row_nodes[c0 + r] * FEAT + dd]
                       : src_conv[(size_t)(c0 + r) * FEAT + dd];
    }
    AggIn[(size_t)r * KP_AGG1 + d] = (bf16_t)val;
  }
}

// ------------------------------ host driver --------------------------------
extern "C" void kernel_launch(void* const* d_in, const int* in_sizes, int n_in,
                              void* d_out, int out_size, void* d_ws, size_t ws_size,
                              hipStream_t stream) {
  (void)in_sizes; (void)n_in; (void)out_size; (void)ws_size;
  const int*   src_idx  = (const int*)  d_in[0];
  const int*   tgt_idx  = (const int*)  d_in[1];
  const float* cut_time = (const float*)d_in[2];
  const int*   nghN     = (const int*)  d_in[3];
  const int*   nghE     = (const int*)  d_in[4];
  const float* nghT     = (const float*)d_in[5];
  const float* n_feat   = (const float*)d_in[6];
  const float* e_feat   = (const float*)d_in[7];
  const float* basis    = (const float*)d_in[8];
  const float* phase    = (const float*)d_in[9];
  const float* w_qs     = (const float*)d_in[10];
  const float* w_ks     = (const float*)d_in[11];
  const float* w_vs     = (const float*)d_in[12];
  const float* fc_w     = (const float*)d_in[13];
  const float* agg1     = (const float*)d_in[14];
  const float* agg2     = (const float*)d_in[15];
  const float* gamma    = (const float*)d_in[16];
  const float* beta     = (const float*)d_in[17];
  float* out = (float*)d_out;

  // bump allocator over d_ws (total ~55 MB)
  char* ws = (char*)d_ws; size_t off = 0;
  auto alloc = [&](size_t bytes) -> void* {
    void* p = ws + off; off = (off + bytes + 255) & ~(size_t)255; return p;
  };
  bf16_t* wb_q  = (bf16_t*)alloc((size_t)NP_QKV * KP_QKV * sizeof(bf16_t));
  bf16_t* wb_k  = (bf16_t*)alloc((size_t)NP_QKV * KP_QKV * sizeof(bf16_t));
  bf16_t* wb_v  = (bf16_t*)alloc((size_t)NP_QKV * KP_QKV * sizeof(bf16_t));
  bf16_t* wb_fc = (bf16_t*)alloc((size_t)NP_QKV * KP_QKV * sizeof(bf16_t));
  bf16_t* wb_a1 = (bf16_t*)alloc((size_t)NP_AGG * KP_AGG1 * sizeof(bf16_t));
  bf16_t* wb_a2 = (bf16_t*)alloc((size_t)NP_AGG * KP_AGG2 * sizeof(bf16_t));
  int*    n1_nodes = (int*)  alloc((size_t)BATCH * NNEI * sizeof(int));
  float*  n1_ts    = (float*)alloc((size_t)BATCH * NNEI * sizeof(float));
  float*  h1_n1    = (float*)alloc((size_t)BATCH * NNEI * FEAT * sizeof(float));
  float*  h1_src   = (float*)alloc((size_t)BATCH * FEAT * sizeof(float));
  // per-chunk pipeline buffers
  bf16_t* Kin   = (bf16_t*)alloc((size_t)CHUNK * NNEI * KP_QKV * sizeof(bf16_t));
  bf16_t* Qin   = (bf16_t*)alloc((size_t)CHUNK * KP_QKV * sizeof(bf16_t));
  float*  khb   = (float*) alloc((size_t)CHUNK * NNEI * MODEL * sizeof(float));
  float*  vhb   = (float*) alloc((size_t)CHUNK * NNEI * MODEL * sizeof(float));
  float*  qhb   = (float*) alloc((size_t)CHUNK * MODEL * sizeof(float));
  bf16_t* obf   = (bf16_t*)alloc((size_t)CHUNK * KP_QKV * sizeof(bf16_t));
  float*  ofc   = (float*) alloc((size_t)CHUNK * MODEL * sizeof(float));
  bf16_t* aggin = (bf16_t*)alloc((size_t)CHUNK * KP_AGG1 * sizeof(bf16_t));
  bf16_t* relub = (bf16_t*)alloc((size_t)CHUNK * KP_AGG2 * sizeof(bf16_t));

  auto pack = [&](const float* W, bf16_t* Wb, int rows, int cols, int rowsP, int colsP) {
    dim3 g((colsP + 255) / 256, rowsP);
    k_pack_weight<<<g, 256, 0, stream>>>(W, Wb, rows, cols, rowsP, colsP);
  };
  pack(w_qs, wb_q,  MODEL, MODEL, NP_QKV, KP_QKV);
  pack(w_ks, wb_k,  MODEL, MODEL, NP_QKV, KP_QKV);
  pack(w_vs, wb_v,  MODEL, MODEL, NP_QKV, KP_QKV);
  pack(fc_w, wb_fc, MODEL, MODEL, NP_QKV, KP_QKV);
  pack(agg1, wb_a1, FEAT, MODEL + FEAT, NP_AGG, KP_AGG1);
  pack(agg2, wb_a2, FEAT, FEAT,         NP_AGG, KP_AGG2);

  // one attention block (chunked); dst is R x 172 fp32
  auto attn_block = [&](int R, const int* row_nodes, const float* row_ts,
                        const float* conv_rows, int conv_gather,
                        const float* src_conv, int src_gather, float* dst) {
    for (int c0 = 0; c0 < R; c0 += CHUNK) {
      int Rc = (R - c0 < CHUNK) ? (R - c0) : CHUNK;
      int Rk = Rc * NNEI;
      k_build_kv<<<Rk, KP_QKV, 0, stream>>>(row_nodes, row_ts, c0, nghN, nghE, nghT,
                                            n_feat, e_feat, basis, phase,
                                            conv_rows, conv_gather, Kin);
      k_build_q<<<Rc, KP_QKV, 0, stream>>>(src_conv, src_gather, row_nodes,
                                           n_feat, phase, c0, Qin);
      k_gemm_bf16<false><<<dim3(Rc / 32, (NT_QKV + 7) / 8), 256, 0, stream>>>(
          Qin, KP_QKV, Rc, wb_q, nullptr, KP_QKV, NT_QKV,
          qhb, nullptr, MODEL, MODEL, nullptr, 0, 0);
      k_gemm_bf16<true><<<dim3(Rk / 32, (NT_QKV + 7) / 8), 256, 0, stream>>>(
          Kin, KP_QKV, Rk, wb_k, wb_v, KP_QKV, NT_QKV,
          khb, vhb, MODEL, MODEL, nullptr, 0, 0);
      k_attention<<<Rc, 128, 0, stream>>>(qhb, khb, vhb, row_nodes, c0, nghN, obf);
      k_gemm_bf16<false><<<dim3(Rc / 32, (NT_QKV + 7) / 8), 256, 0, stream>>>(
          obf, KP_QKV, Rc, wb_fc, nullptr, KP_QKV, NT_QKV,
          ofc, nullptr, MODEL, MODEL, nullptr, 0, 0);
      k_ln_agg<<<Rc, 256, 0, stream>>>(ofc, src_conv, src_gather, row_nodes,
                                       n_feat, phase, gamma, beta, c0, aggin);
      size_t nz = (size_t)Rc * KP_AGG2;
      k_zero_bf16<<<(unsigned)((nz + 255) / 256), 256, 0, stream>>>(relub, nz);
      k_gemm_bf16<false><<<dim3(Rc / 32, (NT_AGG + 7) / 8), 256, 0, stream>>>(
          aggin, KP_AGG1, Rc, wb_a1, nullptr, KP_AGG1, NT_AGG,
          nullptr, nullptr, 0, FEAT, relub, KP_AGG2, 1);
      k_gemm_bf16<false><<<dim3(Rc / 32, (NT_AGG + 7) / 8), 256, 0, stream>>>(
          relub, KP_AGG2, Rc, wb_a2, nullptr, KP_AGG2, NT_AGG,
          dst + (size_t)c0 * FEAT, nullptr, FEAT, FEAT, nullptr, 0, 0);
    }
  };

  auto tem_conv2 = [&](const int* idx, float* dst) {
    k_gather_rows<<<(BATCH * NNEI + 255) / 256, 256, 0, stream>>>(
        idx, nghN, nghT, n1_nodes, n1_ts, BATCH);
    // layer-1 over the 8192 first-hop neighbors (conv & src gathered from n_feat)
    attn_block(BATCH * NNEI, n1_nodes, n1_ts, nullptr, 1, nullptr, 1, h1_n1);
    // layer-1 over the 512 source nodes
    attn_block(BATCH, idx, cut_time, nullptr, 1, nullptr, 1, h1_src);
    // layer-2: neighbor conv = h1_n1 (linear rows), src conv = h1_src
    attn_block(BATCH, idx, cut_time, h1_n1, 0, h1_src, 0, dst);
  };

  tem_conv2(src_idx, out);
  tem_conv2(tgt_idx, out + (size_t)BATCH * FEAT);
}
